// LSTMEncoder_78889959292935
// MI455X (gfx1250) — matrix operand order repaired and checked by reading, not compiled, not run on hardware
//
#include <hip/hip_runtime.h>

// ---------------------------------------------------------------------------
// 2-layer LSTM encoder for MI455X (gfx1250, wave32, WMMA bf16 16x16x32).
// B=64, T=512, IN=128, H=512.
//  - Input projections: big bf16 WMMA GEMMs (parallel over B*T).
//  - Recurrent chain: ONE persistent kernel per layer (128 blocks, 1 wave
//    each); 512 timesteps iterated in-kernel with a device-wide counter
//    barrier per step.  W_hh column slice cached in LDS (64 KB/block),
//    cell state c held in VGPRs for the whole sequence.
// ---------------------------------------------------------------------------

#define DEV __device__ __forceinline__

static constexpr int Bb = 64;
static constexpr int Tt = 512;
static constexpr int IN = 128;
static constexpr int Hh = 512;
static constexpr int N4 = 4 * Hh;         // 2048
static constexpr int Mm = Bb * Tt;        // 32768
static constexpr int BH = Bb * Hh;        // 32768
static constexpr int KT_H = Hh / 32;      // 16 k-tiles over H

typedef __attribute__((ext_vector_type(16))) __bf16        v16bf;
typedef __attribute__((ext_vector_type(8)))  float         v8f;
typedef __attribute__((ext_vector_type(4)))  unsigned int  u32x4;

union FragU {
  u32x4 q[2];
  v16bf v;
};

DEV unsigned short f2bf(float x) {
  union { float f; unsigned int u; } cv; cv.f = x;
  unsigned int u = cv.u;
  unsigned int r = u + 0x7FFFu + ((u >> 16) & 1u);   // round-to-nearest-even
  return (unsigned short)(r >> 16);
}

// A fragment (16x32 bf16, row-major source, leading dim = ld):
// lane L: row m = mBase + (L&15); elems 0..15 = k0..k0+15,
// k0 = kBase + 16*(L>>4).   (contiguous k order, matches packed B)
// NOTE the two chunks are contiguous.
DEV v16bf load_a_frag(const unsigned short* A, int ld, int mBase, int kBase) {
  int lane = threadIdx.x & 31;
  int m  = mBase + (lane & 15);
  int k0 = kBase + ((lane >> 4) << 4);
  const unsigned short* p = A + (size_t)m * ld + k0;
  FragU f;
  f.q[0] = *(const u32x4*)(p);
  f.q[1] = *(const u32x4*)(p + 8);
  return f.v;
}

// B fragment from pre-packed weights: 16 CONTIGUOUS u16 (32B) per lane.
DEV v16bf load_b_frag(const unsigned short* Wp, int KT, int ntile, int kt) {
  int lane = threadIdx.x & 31;
  const unsigned short* p = Wp + ((((size_t)ntile * KT + kt) * 32 + lane) << 4);
  FragU f;
  f.q[0] = *(const u32x4*)(p);
  f.q[1] = *(const u32x4*)(p + 8);       // contiguous (bugfix: was +16)
  return f.v;
}

// Device-wide split barrier for the persistent layer kernels.
DEV void grid_barrier(unsigned int* ctr, unsigned int target) {
  __threadfence();                       // release h writes to device scope
  if ((threadIdx.x & 31) == 0) {
    __hip_atomic_fetch_add(ctr, 1u, __ATOMIC_ACQ_REL, __HIP_MEMORY_SCOPE_AGENT);
    while (__hip_atomic_load(ctr, __ATOMIC_ACQUIRE, __HIP_MEMORY_SCOPE_AGENT) < target) {
      __builtin_amdgcn_s_sleep(1);
    }
  }
  __threadfence();                       // acquire: discard stale lines
}

// ---------------------------------------------------------------------------
// Utility kernels
// ---------------------------------------------------------------------------

__global__ void zero_u32_kernel(unsigned int* p, int n) {
  int i = blockIdx.x * 256 + threadIdx.x;
  if (i < n) p[i] = 0u;
}

__global__ void cvt_bf16_kernel(const float* __restrict__ src,
                                unsigned short* __restrict__ dst, int n) {
  int i = blockIdx.x * 256 + threadIdx.x;
  if (i < n) dst[i] = f2bf(src[i]);
}

// Pack W [rows = N4out, cols = K] (row-major f32, gates = act . W^T) into
// WMMA-B-fragment order (lane = col L&15, k = e + 16*(L>>4); 16 u16
// contiguous per lane).
__global__ void pack_w_kernel(const float* __restrict__ W,
                              unsigned short* __restrict__ Wp,
                              int N4out, int K) {
  int idx = blockIdx.x * 256 + threadIdx.x;
  int total = N4out * K;
  if (idx >= total) return;
  int KT   = K >> 5;
  int e    = idx & 15;
  int lane = (idx >> 4) & 31;
  int tile = idx >> 9;
  int kt   = tile % KT;
  int nt   = tile / KT;
  int n = nt * 16 + (lane & 15);
  int k = kt * 32 + ((lane >> 4) << 4) + e;
  Wp[idx] = f2bf(W[(size_t)n * K + k]);
}

// ---------------------------------------------------------------------------
// Input-projection GEMM: xw[t, b, n] = sum_k A[b*T+t, k]*W[n, k] + b1[n]+b2[n]
// A is row-major [B*T, K] bf16.  xw stored t-major so each recurrent step
// reads one contiguous 512KB slab.
// ---------------------------------------------------------------------------

__global__ void __launch_bounds__(256)
gemm_xw_kernel(const unsigned short* __restrict__ Abf,
               const unsigned short* __restrict__ Wp,
               const float* __restrict__ b1, const float* __restrict__ b2,
               float* __restrict__ xw, int K) {
  const int nGroups = N4 / 64;                       // 32
  int unit = blockIdx.x * blockDim.y + threadIdx.y;  // 0 .. (M/16)*32 - 1
  int mt = unit / nGroups;
  int ng = unit % nGroups;
  int KT = K >> 5;

  v8f acc[4] = {};
  for (int kt = 0; kt < KT; ++kt) {
    v16bf a = load_a_frag(Abf, K, mt * 16, kt * 32);
#pragma unroll
    for (int j = 0; j < 4; ++j) {
      v16bf b = load_b_frag(Wp, KT, ng * 4 + j, kt);
      acc[j] = __builtin_amdgcn_wmma_f32_16x16x32_bf16(
          false, a, false, b, (short)0, acc[j], false, false);
    }
  }

  int lane  = threadIdx.x & 31;
  int mBase = mt * 16 + ((lane >> 4) << 3);
  int nLane = lane & 15;
#pragma unroll
  for (int j = 0; j < 4; ++j) {
    int n = ng * 64 + j * 16 + nLane;
    float bias = b1[n] + b2[n];
#pragma unroll
    for (int r = 0; r < 8; ++r) {
      int m  = mBase + r;                // global row = b*T + t
      int bb = m >> 9;                   // /T
      int tt = m & (Tt - 1);             // %T
      xw[((size_t)tt * Bb + bb) * N4 + n] = acc[j][r] + bias;
    }
  }
}

// ---------------------------------------------------------------------------
// Persistent recurrent layer.  Grid (H/16=32, B/16=4) = 128 blocks, 1 wave
// each.  Each block owns a 16x16 (batch x hidden) tile: 4 gate accumulators,
// c kept in VGPRs for all 512 steps, and its entire W_hh column slice
// (4 gates x 16 k-tiles x 1KB = 64KB) cached in LDS, so the per-step inner
// loop reads weights via ds_load_b128 and only h via global loads.
// ---------------------------------------------------------------------------

__global__ void __launch_bounds__(32)
lstm_layer_kernel(const float* __restrict__ xw,          // [T, B, 4H] f32
                  const unsigned short* __restrict__ Whhp,
                  unsigned short* __restrict__ hbf,      // 2*BH ping-pong
                  unsigned short* __restrict__ out_bf,   // layer0 seq out (or null)
                  float* __restrict__ out_f,             // layer1 seq out (or null)
                  float* __restrict__ hn_dst,            // [B,H] final h
                  float* __restrict__ cn_dst,            // [B,H] final c
                  unsigned int* __restrict__ ctr) {
  extern __shared__ u32x4 wlds[];         // 64KB: (4*KT_H) frags * 32 lanes * 32B
  const int nBlocks = (Hh / 16) * (Bb / 16);  // 128
  int nt   = blockIdx.x;                  // hidden 16-tile
  int mt   = blockIdx.y;                  // batch tile
  int lane = threadIdx.x & 31;
  int half = lane >> 4;

  // ---- one-time: stage this block's W_hh slice into LDS -------------------
  for (int s = 0; s < 4 * KT_H; ++s) {
    int g  = s / KT_H;
    int kt = s % KT_H;
    const u32x4* src = (const u32x4*)(
        Whhp + ((((size_t)(g * (Hh / 16) + nt) * KT_H + kt) * 32 + lane) << 4));
    int d = (s * 32 + lane) * 2;
    wlds[d + 0] = src[0];
    wlds[d + 1] = src[1];
  }

  float creg[8];
#pragma unroll
  for (int r = 0; r < 8; ++r) creg[r] = 0.0f;

  for (int t = 0; t < Tt; ++t) {
    const unsigned short* h_in  = hbf + (size_t)(t & 1) * BH;
    unsigned short*       h_out = hbf + (size_t)((t + 1) & 1) * BH;

    v8f acc[4] = {};
    for (int kt = 0; kt < KT_H; ++kt) {
      v16bf a = load_a_frag(h_in, Hh, mt * 16, kt * 32);
#pragma unroll
      for (int g = 0; g < 4; ++g) {
        int d = ((g * KT_H + kt) * 32 + lane) * 2;
        FragU f;
        f.q[0] = wlds[d + 0];
        f.q[1] = wlds[d + 1];
        acc[g] = __builtin_amdgcn_wmma_f32_16x16x32_bf16(
            false, a, false, f.v, (short)0, acc[g], false, false);
      }
    }

    const float* xwt = xw + (size_t)t * Bb * N4;
    int n = nt * 16 + (lane & 15);
#pragma unroll
    for (int r = 0; r < 8; ++r) {
      int b = mt * 16 + half * 8 + r;
      size_t row = (size_t)b * N4;
      float gi = acc[0][r] + xwt[row + 0 * Hh + n];
      float gf = acc[1][r] + xwt[row + 1 * Hh + n];
      float gg = acc[2][r] + xwt[row + 2 * Hh + n];
      float go = acc[3][r] + xwt[row + 3 * Hh + n];
      float si = 1.0f / (1.0f + __expf(-gi));
      float sf = 1.0f / (1.0f + __expf(-gf));
      float so = 1.0f / (1.0f + __expf(-go));
      float tg = tanhf(gg);
      float cv = sf * creg[r] + si * tg;
      float hv = so * tanhf(cv);
      creg[r] = cv;
      h_out[(size_t)b * Hh + n] = f2bf(hv);
      size_t oidx = ((size_t)b * Tt + t) * Hh + n;
      if (out_bf) out_bf[oidx] = f2bf(hv);
      if (out_f)  out_f[oidx]  = hv;
      if (t == Tt - 1) {
        hn_dst[(size_t)b * Hh + n] = hv;
        cn_dst[(size_t)b * Hh + n] = cv;
      }
    }

    grid_barrier(ctr, (unsigned)(nBlocks * (t + 1)));
  }
}

// ---------------------------------------------------------------------------
// Host-side orchestration
// ---------------------------------------------------------------------------

extern "C" void kernel_launch(void* const* d_in, const int* in_sizes, int n_in,
                              void* d_out, int out_size, void* d_ws, size_t ws_size,
                              hipStream_t stream) {
  (void)in_sizes; (void)n_in; (void)out_size; (void)ws_size;

  const float* x    = (const float*)d_in[0];
  const float* Wih0 = (const float*)d_in[1];
  const float* Whh0 = (const float*)d_in[2];
  const float* bih0 = (const float*)d_in[3];
  const float* bhh0 = (const float*)d_in[4];
  const float* Wih1 = (const float*)d_in[5];
  const float* Whh1 = (const float*)d_in[6];
  const float* bih1 = (const float*)d_in[7];
  const float* bhh1 = (const float*)d_in[8];
  float* out = (float*)d_out;
  const size_t OUT1 = (size_t)Bb * Tt * Hh;           // 16,777,216

  // ---- workspace layout --------------------------------------------------
  char* ws = (char*)d_ws;
  size_t off = 0;
  float* xw = (float*)(ws + off);                      off += (size_t)Mm * N4 * 4; // 256 MB (reused)
  unsigned short* xbf   = (unsigned short*)(ws + off); off += (size_t)Mm * IN * 2;
  unsigned short* out0b = (unsigned short*)(ws + off); off += (size_t)Mm * Hh * 2;
  unsigned short* wih0p = (unsigned short*)(ws + off); off += (size_t)N4 * IN * 2;
  unsigned short* whh0p = (unsigned short*)(ws + off); off += (size_t)N4 * Hh * 2;
  unsigned short* wih1p = (unsigned short*)(ws + off); off += (size_t)N4 * Hh * 2;
  unsigned short* whh1p = (unsigned short*)(ws + off); off += (size_t)N4 * Hh * 2;
  char* stateBase = ws + off;
  unsigned int*   ctrs = (unsigned int*)stateBase;     // [2], padded to 256B
  unsigned short* hbf0 = (unsigned short*)(stateBase + 256);   // 2*BH u16
  unsigned short* hbf1 = hbf0 + 2 * BH;                        // 2*BH u16
  const int stateU32 = (256 + 4 * BH * 2) / 4;         // 65,600 dwords

  // ---- re-init per call (determinism under graph replay) ------------------
  zero_u32_kernel<<<(stateU32 + 255) / 256, 256, 0, stream>>>(
      (unsigned int*)stateBase, stateU32);

  // ---- convert x, pack weights into WMMA fragment order -------------------
  cvt_bf16_kernel<<<(Mm * IN + 255) / 256, 256, 0, stream>>>(x, xbf, Mm * IN);
  pack_w_kernel<<<(N4 * IN + 255) / 256, 256, 0, stream>>>(Wih0, wih0p, N4, IN);
  pack_w_kernel<<<(N4 * Hh + 255) / 256, 256, 0, stream>>>(Whh0, whh0p, N4, Hh);
  pack_w_kernel<<<(N4 * Hh + 255) / 256, 256, 0, stream>>>(Wih1, wih1p, N4, Hh);
  pack_w_kernel<<<(N4 * Hh + 255) / 256, 256, 0, stream>>>(Whh1, whh1p, N4, Hh);

  const int gemmUnits = (Mm / 16) * (N4 / 64);         // 65536 waves
  const dim3 gemmBlock(32, 8);
  const int  gemmGrid = gemmUnits / 8;                 // 8192 blocks
  const dim3 layerGrid(Hh / 16, Bb / 16);              // (32, 4) = 128 blocks
  const size_t layerLds = (size_t)4 * KT_H * 32 * 32;  // 64 KB

  // ---- layer 0 ------------------------------------------------------------
  gemm_xw_kernel<<<gemmGrid, gemmBlock, 0, stream>>>(xbf, wih0p, bih0, bhh0, xw, IN);
  lstm_layer_kernel<<<layerGrid, 32, layerLds, stream>>>(
      xw, whh0p, hbf0, out0b, nullptr,
      out + OUT1 + 0 * BH,            // h_n[0]
      out + OUT1 + 2 * BH,            // c_n[0]
      ctrs + 0);

  // ---- layer 1 ------------------------------------------------------------
  gemm_xw_kernel<<<gemmGrid, gemmBlock, 0, stream>>>(out0b, wih1p, bih1, bhh1, xw, Hh);
  lstm_layer_kernel<<<layerGrid, 32, layerLds, stream>>>(
      xw, whh1p, hbf1, nullptr, out,
      out + OUT1 + 1 * BH,            // h_n[1]
      out + OUT1 + 3 * BH,            // c_n[1]
      ctrs + 1);
}
